// SSA_10582799418112
// MI455X (gfx1250) — matrix operand order repaired and checked by reading, not compiled, not run on hardware
//
#include <hip/hip_runtime.h>
#include <hip/hip_bf16.h>

// ---------------- problem constants ----------------
#define Bn    2
#define Cc    256
#define Hh    128
#define Ww    128
#define HWc   (Hh * Ww)          // 16384
#define RHO   2
#define WINc  5
#define HEADS 8
#define Dh    32                 // C / HEADS
#define Hg    64
#define Wg    64
#define NPIX  (Bn * HWc)         // 32768 pixels
#define NROW  (4 * Bn * Hg * Wg) // 32768 sampled rows (4 subgrids)

typedef __bf16 bf16_t;
typedef __attribute__((ext_vector_type(16))) __bf16 v16bf;
typedef __attribute__((ext_vector_type(8)))  float  v8f;
typedef int vi4 __attribute__((vector_size(16)));   // matches builtin's v4i pointee

static __device__ __forceinline__ unsigned short f2bf(float f) {
  unsigned u = __float_as_uint(f);
  unsigned r = (u + 0x7FFFu + ((u >> 16) & 1u)) >> 16;
  return (unsigned short)r;
}
static __device__ __forceinline__ float bf2f(unsigned short h) {
  return __uint_as_float(((unsigned)h) << 16);
}
static __device__ __forceinline__ float gelu_exact(float v) {
  return 0.5f * v * (1.0f + erff(v * 0.70710678118654752f));
}

// ---- CDNA5 async global->LDS copy (ASYNCcnt path), with safe fallback ----
static __device__ __forceinline__ void async_copy16(const bf16_t* gsrc, bf16_t* ldst) {
#if __has_builtin(__builtin_amdgcn_global_load_async_to_lds_b128)
  __builtin_amdgcn_global_load_async_to_lds_b128(
      (__attribute__((address_space(1))) vi4*)(void*)(const_cast<bf16_t*>(gsrc)),
      (__attribute__((address_space(3))) vi4*)(void*)(ldst), 0, 0);
#else
  *(uint4*)ldst = *(const uint4*)gsrc;
#endif
}
static __device__ __forceinline__ void async_wait0() {
#if __has_builtin(__builtin_amdgcn_s_wait_asynccnt)
  __builtin_amdgcn_s_wait_asynccnt(0);
#else
  asm volatile("s_wait_asynccnt 0" ::: "memory");
#endif
}

// ---------------- fp32 -> bf16 weight conversion ----------------
__global__ void cvt_bf16_kernel(const float* __restrict__ in,
                                unsigned short* __restrict__ out, int n) {
  int i = blockIdx.x * 256 + threadIdx.x;
  if (i < n) out[i] = f2bf(in[i]);
}

// ---------------- LayerNorm over C: NCHW in -> NHWC out ----------------
template <bool OUT_BF16>
__global__ __launch_bounds__(256) void ln_kernel(const float* __restrict__ x,
                                                 const float* __restrict__ gw,
                                                 const float* __restrict__ bw,
                                                 void* __restrict__ outp) {
  __shared__ float tile[Cc][33];   // padded: conflict-free both phases
  __shared__ float psum[8][32];
  __shared__ float psq[8][32];
  __shared__ float mu[32];
  __shared__ float rs[32];
  const int t = threadIdx.x;
  const int lane = t & 31;
  const int chunk = t >> 5;                     // one wave per chunk
  const long p0 = (long)blockIdx.x * 32;
  const int b = (int)(p0 / HWc);
  const int hw0 = (int)(p0 % HWc);
  float s = 0.f, sq = 0.f;
  for (int i = 0; i < 32; ++i) {
    int c = chunk * 32 + i;
    float v = x[((long)(b * Cc + c)) * HWc + hw0 + lane];  // lanes coalesced
    tile[c][lane] = v;
    s += v; sq += v * v;
  }
  psum[chunk][lane] = s;
  psq[chunk][lane] = sq;
  __syncthreads();
  if (t < 32) {
    float S = 0.f, Q = 0.f;
    for (int k = 0; k < 8; ++k) { S += psum[k][t]; Q += psq[k][t]; }
    float m = S * (1.0f / Cc);
    float var = Q * (1.0f / Cc) - m * m;
    mu[t] = m;
    rs[t] = rsqrtf(var + 1e-5f);
  }
  __syncthreads();
  const float g = gw[t], bb = bw[t];
  for (int i = 0; i < 32; ++i) {
    float v = (tile[t][i] - mu[i]) * rs[i] * g + bb;
    long o = (p0 + i) * Cc + t;                 // lanes vary channel: coalesced
    if (OUT_BF16) ((unsigned short*)outp)[o] = f2bf(v);
    else          ((float*)outp)[o] = v;
  }
}

// ---------------- depthwise 3x3 (pad 1), NHWC ----------------
template <bool GELU, bool OUT_BF16>
__global__ __launch_bounds__(256) void dw3x3_kernel(const float* __restrict__ in,
                                                    const float* __restrict__ w,
                                                    const float* __restrict__ bias,
                                                    void* __restrict__ outp,
                                                    int IH, int IW) {
  const int c = threadIdx.x;
  const long pix = blockIdx.x;
  const int xg = (int)(pix % IW);
  long tmp = pix / IW;
  const int yg = (int)(tmp % IH);
  const long img = tmp / IH;
  float acc = bias[c];
  const float* wc = w + c * 9;
#pragma unroll
  for (int dy = -1; dy <= 1; ++dy)
#pragma unroll
    for (int dx = -1; dx <= 1; ++dx) {
      int yy = yg + dy, xx = xg + dx;
      if (yy >= 0 && yy < IH && xx >= 0 && xx < IW)
        acc += wc[(dy + 1) * 3 + (dx + 1)] *
               in[((img * IH + yy) * (long)IW + xx) * Cc + c];
    }
  if (GELU) acc = gelu_exact(acc);
  long o = pix * Cc + c;
  if (OUT_BF16) ((unsigned short*)outp)[o] = f2bf(acc);
  else          ((float*)outp)[o] = acc;
}

// ---------------- 1x1 conv C->2 (offset head) ----------------
__global__ __launch_bounds__(256) void offpw_kernel(const float* __restrict__ in,
                                                    const float* __restrict__ w,
                                                    const float* __restrict__ b,
                                                    float* __restrict__ out) {
  long pix = (long)blockIdx.x * 256 + threadIdx.x;
  const float* row = in + pix * Cc;
  float a0 = b[0], a1 = b[1];
  for (int c = 0; c < Cc; c += 4) {
    float4 v = *(const float4*)(row + c);
    a0 += v.x * w[c] + v.y * w[c + 1] + v.z * w[c + 2] + v.w * w[c + 3];
    a1 += v.x * w[Cc + c] + v.y * w[Cc + c + 1] + v.z * w[Cc + c + 2] + v.w * w[Cc + c + 3];
  }
  out[pix * 2 + 0] = a0;
  out[pix * 2 + 1] = a1;
}

// ---------------- deformable bilinear sample (border), all 4 subgrids ----------------
__global__ __launch_bounds__(256) void sample_kernel(const float* __restrict__ xn,
                                                     const float* __restrict__ off,
                                                     float* __restrict__ g) {
  const int c = threadIdx.x;
  const long row = blockIdx.x;
  const int xg = (int)(row % Wg);
  long t = row / Wg;
  const int yg = (int)(t % Hg);
  t /= Hg;
  const int b = (int)(t % Bn);
  const int sub = (int)(t / Bn);
  const int kk = sub >> 1, ll = sub & 1;
  const int yi = yg * RHO + kk, xi = xg * RHO + ll;
  const long opix = ((long)b * Hh + yi) * Ww + xi;
  const float o0 = tanhf(off[opix * 2 + 0]) * 2.0f;
  const float o1 = tanhf(off[opix * 2 + 1]) * 2.0f;
  float ys = fminf(fmaxf((float)yi + o0, 0.f), (float)(Hh - 1));
  float xs = fminf(fmaxf((float)xi + o1, 0.f), (float)(Ww - 1));
  float y0 = floorf(ys), x0 = floorf(xs);
  float wy = ys - y0, wx = xs - x0;
  int y0i = min(max((int)y0, 0), Hh - 1), y1i = min(y0i + 1, Hh - 1);
  int x0i = min(max((int)x0, 0), Ww - 1), x1i = min(x0i + 1, Ww - 1);
  const float* base = xn + (long)b * HWc * Cc;
  float v00 = base[((long)y0i * Ww + x0i) * Cc + c];
  float v01 = base[((long)y0i * Ww + x1i) * Cc + c];
  float v10 = base[((long)y1i * Ww + x0i) * Cc + c];
  float v11 = base[((long)y1i * Ww + x1i) * Cc + c];
  g[row * Cc + c] = v00 * (1 - wy) * (1 - wx) + v01 * (1 - wy) * wx +
                    v10 * wy * (1 - wx) + v11 * wy * wx;
}

// ---------------- weight-stationary bf16 WMMA GEMM ----------------
// Y[M,N] = act(X[M,K] * W[N,K]^T + bias).
// Block = 8 waves: stages a 64-column weight strip (K-chunked, 32KB) into LDS via
// async-to-LDS, then each wave computes MIT 16x64 output strips against it.
// Fragment layouts follow CDNA5 ISA 7.12.2 (16-bit A 16x32, B 32x16, f32 C 16x16).
// OUTMODE: 0 = bf16 row-major, 1 = f32 row-major, 2 = f32 NCHW scatter + residual add.
template <int ACT, int OUTMODE, int MIT>
__global__ __launch_bounds__(256) void gemm_kernel(const unsigned short* __restrict__ Xu,
                                                   const unsigned short* __restrict__ Wu,
                                                   const float* __restrict__ bias,
                                                   void* __restrict__ Yp,
                                                   const float* __restrict__ resid,
                                                   int M, int Nn, int Kk) {
  constexpr int KC = 256;           // K staged per chunk
  constexpr int LDC = KC + 8;       // pad 8 bf16 -> ds_load_b128 at 2-way floor
  __shared__ bf16_t wtile[64 * LDC];          // 33792 B
  const int wave = threadIdx.x >> 5;
  const int lane = threadIdx.x & 31;
  const int ln = lane & 15;
  const int hi = lane >> 4;         // wave-uniform usage only; EXEC stays full
  const int n0 = blockIdx.x * 64;
  const int mbase = blockIdx.y * (128 * MIT);
  const bf16_t* X = (const bf16_t*)Xu;
  const bf16_t* Wm = (const bf16_t*)Wu;
  v8f acc[MIT][4];
#pragma unroll
  for (int m = 0; m < MIT; ++m)
#pragma unroll
    for (int j = 0; j < 4; ++j)
#pragma unroll
      for (int r = 0; r < 8; ++r) acc[m][j][r] = 0.f;

  for (int kc = 0; kc < Kk; kc += KC) {
    if (kc) __syncthreads();        // previous chunk fully consumed before overwrite
    // ---- stage 64 x KC weights: 2048 16B units, 8 per thread, wave-coalesced ----
#pragma unroll
    for (int i = 0; i < 8; ++i) {
      int u = threadIdx.x + 256 * i;
      int col = u >> 5;             // 32 units per column
      int kid = (u & 31) * 8;
      async_copy16(Wm + (long)(n0 + col) * Kk + kc + kid, &wtile[col * LDC + kid]);
    }
    async_wait0();
    __syncthreads();
    // ---- compute: MIT m-tiles x (KC/32) k-steps, B from LDS, A streamed ----
#pragma unroll
    for (int m = 0; m < MIT; ++m) {
      const int m0 = mbase + (m * 8 + wave) * 16;
      // A: lane<16 -> row (m0+ln), K [k0..k0+7]+[k0+16..k0+23]; lane>=16 shifts by 8
      const bf16_t* arow = X + (long)(m0 + ln) * Kk + kc + hi * 8;
      for (int k0 = 0; k0 < KC; k0 += 32) {
        v16bf a;
        *((uint4*)&a)       = *((const uint4*)(arow + k0));
        *(((uint4*)&a) + 1) = *((const uint4*)(arow + k0 + 16));
#pragma unroll
        for (int j = 0; j < 4; ++j) {
          // B: lane<16 -> column (n0+16j+ln), K [k0..k0+7]+[k0+16..k0+23]; lane>=16 shifts by 8
          const bf16_t* bsrc = &wtile[(j * 16 + ln) * LDC + k0 + hi * 8];
          v16bf bf;
          *((uint4*)&bf)       = *((const uint4*)bsrc);
          *(((uint4*)&bf) + 1) = *((const uint4*)(bsrc + 16));
          acc[m][j] = __builtin_amdgcn_wmma_f32_16x16x32_bf16(false, a, false, bf,
                                                              (short)0, acc[m][j],
                                                              false, false);
        }
      }
    }
  }
  // ---- epilogue ----
#pragma unroll
  for (int m = 0; m < MIT; ++m) {
    const int m0 = mbase + (m * 8 + wave) * 16;
#pragma unroll
    for (int j = 0; j < 4; ++j)
#pragma unroll
      for (int r = 0; r < 8; ++r) {
        int row = m0 + r + hi * 8;    // C/D: VGPR r -> M=r (lanes 0-15) / M=r+8
        int col = n0 + j * 16 + ln;
        float v = acc[m][j][r] + bias[col];
        if (ACT == 1) v = gelu_exact(v);
        if (OUTMODE == 0) {
          ((unsigned short*)Yp)[(long)row * Nn + col] = f2bf(v);
        } else if (OUTMODE == 1) {
          ((float*)Yp)[(long)row * Nn + col] = v;
        } else {
          int b = row >> 14;          // row = b*HW + hw (HW = 16384)
          int hw = row & (HWc - 1);
          long idx = ((long)(b * Cc + col)) * HWc + hw;
          ((float*)Yp)[idx] = v + resid[idx];
        }
      }
  }
}

// ---------------- local 5x5 MHA (zero-padded unfold semantics) + scatter + residual ----
__global__ __launch_bounds__(256) void attn_kernel(const unsigned short* __restrict__ qb,
                                                   const unsigned short* __restrict__ kb,
                                                   const unsigned short* __restrict__ vb,
                                                   const float* __restrict__ x,
                                                   float* __restrict__ xres) {
  const long tid = (long)blockIdx.x * 256 + threadIdx.x;
  const int head = (int)(tid & 7);
  const long row = tid >> 3;
  const int xg = (int)(row & 63);
  long t = row >> 6;
  const int yg = (int)(t & 63);
  t >>= 6;
  const int b = (int)(t & (Bn - 1));
  const int sub = (int)(t >> 1);
  const long imgbase = (row >> 12) << 12;  // (sub*B+b) * 4096
  float q[Dh];
  const unsigned short* qp = qb + row * Cc + head * Dh;
#pragma unroll
  for (int i = 0; i < Dh; ++i) q[i] = bf2f(qp[i]);
  float sc[WINc * WINc];
  const float scale = 0.1767766952966369f;  // 1/sqrt(32)
#pragma unroll
  for (int ky = 0; ky < WINc; ++ky)
#pragma unroll
    for (int kx = 0; kx < WINc; ++kx) {
      int yy = yg + ky - 2, xx = xg + kx - 2;
      float s = 0.f;                        // OOB neighbor: k = 0 -> logit 0 (unfold pad)
      if (yy >= 0 && yy < Hg && xx >= 0 && xx < Wg) {
        const unsigned short* kp = kb + (imgbase + yy * Wg + xx) * Cc + head * Dh;
#pragma unroll
        for (int i = 0; i < Dh; ++i) s += q[i] * bf2f(kp[i]);
        s *= scale;
      }
      sc[ky * WINc + kx] = s;
    }
  float mx = sc[0];
#pragma unroll
  for (int i = 1; i < WINc * WINc; ++i) mx = fmaxf(mx, sc[i]);
  float den = 0.f;
#pragma unroll
  for (int i = 0; i < WINc * WINc; ++i) { sc[i] = __expf(sc[i] - mx); den += sc[i]; }
  const float inv = 1.f / den;
  float o[Dh];
#pragma unroll
  for (int i = 0; i < Dh; ++i) o[i] = 0.f;
#pragma unroll
  for (int ky = 0; ky < WINc; ++ky)
#pragma unroll
    for (int kx = 0; kx < WINc; ++kx) {
      int yy = yg + ky - 2, xx = xg + kx - 2;
      if (yy >= 0 && yy < Hg && xx >= 0 && xx < Wg) {
        float a = sc[ky * WINc + kx] * inv;
        const unsigned short* vp = vb + (imgbase + yy * Wg + xx) * Cc + head * Dh;
#pragma unroll
        for (int i = 0; i < Dh; ++i) o[i] += a * bf2f(vp[i]);
      }
    }
  const int kk = sub >> 1, ll = sub & 1;
  const int yi = yg * RHO + kk, xi = xg * RHO + ll;
#pragma unroll
  for (int i = 0; i < Dh; ++i) {
    int c = head * Dh + i;
    long idx = ((long)(b * Cc + c)) * HWc + yi * Ww + xi;
    xres[idx] = x[idx] + o[i];   // every (pixel,channel) written exactly once
  }
}

// ---------------- host-side orchestration ----------------
extern "C" void kernel_launch(void* const* d_in, const int* in_sizes, int n_in,
                              void* d_out, int out_size, void* d_ws, size_t ws_size,
                              hipStream_t stream) {
  const float* x        = (const float*)d_in[0];
  const float* g1       = (const float*)d_in[1];
  const float* b1       = (const float*)d_in[2];
  const float* g2       = (const float*)d_in[3];
  const float* b2       = (const float*)d_in[4];
  const float* off_dw_w = (const float*)d_in[5];
  const float* off_dw_b = (const float*)d_in[6];
  const float* off_pw_w = (const float*)d_in[7];
  const float* off_pw_b = (const float*)d_in[8];
  const float* q_dw_w   = (const float*)d_in[9];
  const float* q_dw_b   = (const float*)d_in[10];
  const float* q_pw_w   = (const float*)d_in[11];
  const float* q_pw_b   = (const float*)d_in[12];
  const float* k_dw_w   = (const float*)d_in[13];
  const float* k_dw_b   = (const float*)d_in[14];
  const float* k_pw_w   = (const float*)d_in[15];
  const float* k_pw_b   = (const float*)d_in[16];
  const float* v_dw_w   = (const float*)d_in[17];
  const float* v_dw_b   = (const float*)d_in[18];
  const float* v_pw_w   = (const float*)d_in[19];
  const float* v_pw_b   = (const float*)d_in[20];
  const float* mlp_w1   = (const float*)d_in[21];
  const float* mlp_b1   = (const float*)d_in[22];
  const float* mlp_w2   = (const float*)d_in[23];
  const float* mlp_b2   = (const float*)d_in[24];
  float* out = (float*)d_out;

  char* ws = (char*)d_ws;
  const size_t SZ_F = (size_t)NPIX * Cc * 4;   // 33.5 MB fp32 activation
  const size_t SZ_H = (size_t)NPIX * Cc * 2;   // 16.8 MB bf16 activation
  size_t cur = 0;
  float*          xn      = (float*)(ws + cur);          cur += SZ_F;
  float*          off_h   = (float*)(ws + cur);          cur += SZ_F;
  float*          offsets = (float*)(ws + cur);          cur += (size_t)NPIX * 2 * 4;
  float*          gbuf    = (float*)(ws + cur);          cur += SZ_F;
  unsigned short* dwq     = (unsigned short*)(ws + cur); cur += SZ_H;
  unsigned short* dwk     = (unsigned short*)(ws + cur); cur += SZ_H;
  unsigned short* dwv     = (unsigned short*)(ws + cur); cur += SZ_H;
  unsigned short* qb      = (unsigned short*)(ws + cur); cur += SZ_H;
  unsigned short* kb      = (unsigned short*)(ws + cur); cur += SZ_H;
  unsigned short* vb      = (unsigned short*)(ws + cur); cur += SZ_H;
  float*          xres    = (float*)(ws + cur);          cur += SZ_F;
  unsigned short* xn2     = (unsigned short*)(ws + cur); cur += SZ_H;
  unsigned short* h1      = (unsigned short*)(ws + cur); cur += (size_t)NPIX * 1024 * 2;
  unsigned short* wqb     = (unsigned short*)(ws + cur); cur += Cc * Cc * 2;
  unsigned short* wkb     = (unsigned short*)(ws + cur); cur += Cc * Cc * 2;
  unsigned short* wvb     = (unsigned short*)(ws + cur); cur += Cc * Cc * 2;
  unsigned short* w1b     = (unsigned short*)(ws + cur); cur += 4 * Cc * Cc * 2;
  unsigned short* w2b     = (unsigned short*)(ws + cur); cur += 4 * Cc * Cc * 2;

  // weights -> bf16
  cvt_bf16_kernel<<<(Cc * Cc + 255) / 256, 256, 0, stream>>>(q_pw_w, wqb, Cc * Cc);
  cvt_bf16_kernel<<<(Cc * Cc + 255) / 256, 256, 0, stream>>>(k_pw_w, wkb, Cc * Cc);
  cvt_bf16_kernel<<<(Cc * Cc + 255) / 256, 256, 0, stream>>>(v_pw_w, wvb, Cc * Cc);
  cvt_bf16_kernel<<<(4 * Cc * Cc + 255) / 256, 256, 0, stream>>>(mlp_w1, w1b, 4 * Cc * Cc);
  cvt_bf16_kernel<<<(4 * Cc * Cc + 255) / 256, 256, 0, stream>>>(mlp_w2, w2b, 4 * Cc * Cc);

  // LN1 -> xn (NHWC f32)
  ln_kernel<false><<<NPIX / 32, 256, 0, stream>>>(x, g1, b1, xn);
  // offset head
  dw3x3_kernel<true, false><<<NPIX, 256, 0, stream>>>(xn, off_dw_w, off_dw_b, off_h, Hh, Ww);
  offpw_kernel<<<NPIX / 256, 256, 0, stream>>>(off_h, off_pw_w, off_pw_b, offsets);
  // deformable sampling, all 4 subgrids
  sample_kernel<<<NROW, 256, 0, stream>>>(xn, offsets, gbuf);
  // depthwise 3x3 of q/k/v (bf16 out)
  dw3x3_kernel<false, true><<<NROW, 256, 0, stream>>>(gbuf, q_dw_w, q_dw_b, dwq, Hg, Wg);
  dw3x3_kernel<false, true><<<NROW, 256, 0, stream>>>(gbuf, k_dw_w, k_dw_b, dwk, Hg, Wg);
  dw3x3_kernel<false, true><<<NROW, 256, 0, stream>>>(gbuf, v_dw_w, v_dw_b, dwv, Hg, Wg);
  // pointwise q/k/v: [32768,256] x [256,256]^T, weights LDS-stationary, MIT=4 (512 rows/block)
  gemm_kernel<0, 0, 4><<<dim3(Cc / 64, NROW / 512), 256, 0, stream>>>(dwq, wqb, q_pw_b, qb, nullptr, NROW, Cc, Cc);
  gemm_kernel<0, 0, 4><<<dim3(Cc / 64, NROW / 512), 256, 0, stream>>>(dwk, wkb, k_pw_b, kb, nullptr, NROW, Cc, Cc);
  gemm_kernel<0, 0, 4><<<dim3(Cc / 64, NROW / 512), 256, 0, stream>>>(dwv, wvb, v_pw_b, vb, nullptr, NROW, Cc, Cc);
  // local MHA + scatter + residual -> xres (NCHW f32)
  attn_kernel<<<(NROW * HEADS) / 256, 256, 0, stream>>>(qb, kb, vb, x, xres);
  // LN2 -> bf16 NHWC
  ln_kernel<true><<<NPIX / 32, 256, 0, stream>>>(xres, g2, b2, xn2);
  // MLP: [32768,256]x[1024,256]^T (+GELU) then [32768,1024]x[256,1024]^T (+residual, NCHW)
  gemm_kernel<1, 0, 4><<<dim3(1024 / 64, NPIX / 512), 256, 0, stream>>>(xn2, w1b, mlp_b1, h1, nullptr, NPIX, 1024, Cc);
  gemm_kernel<0, 2, 2><<<dim3(Cc / 64, NPIX / 256), 256, 0, stream>>>(h1, w2b, mlp_b2, out, xres, NPIX, Cc, 1024);
}
